// FullAttentionLayer_43550968382294
// MI455X (gfx1250) — compile-verified, matrix operand order checked by reading, not run on hardware
//
#include <hip/hip_runtime.h>
#include <hip/hip_bf16.h>

typedef __attribute__((ext_vector_type(16))) _Float16 v16h;
typedef __attribute__((ext_vector_type(8)))  _Float16 v8h;
typedef __attribute__((ext_vector_type(8)))  float    v8f;

#define DM    1024
#define NHD   16
#define DH    64
#define NBAT  2
#define SEQ   2048
#define NH    (NBAT * NHD)    // 32 (n,h) pairs
#define MTOT  (NBAT * SEQ)    // 4096
#define SCORE_SCALE 0.125f    // 1/sqrt(64)

// ---------------------------------------------------------------------------
// Fragment loader: LDS tile stored as [outer][k] with row stride `ldk` halfs
// (ldk*2 bytes must be a multiple of 16). Implements the CDNA5 16-bit
// A-matrix layout: lanes 0-15 hold K {kb+0..7, kb+16..23}, lanes 16-31 hold
// K {kb+8..15, kb+24..31}. B is fed from a transposed LDS tile so the same
// loader serves both operands. Two ds_load_b128 per fragment.
// ---------------------------------------------------------------------------
__device__ __forceinline__ v16h lds_frag(const _Float16* base, int ldk,
                                         int outer, int kbase) {
  const int lane = threadIdx.x & 31;
  const int r  = outer + (lane & 15);
  const int ko = kbase + ((lane & 16) ? 8 : 0);
  const _Float16* p = base + r * ldk + ko;
  v8h lo = *(const v8h*)(p);        // K = ko .. ko+7
  v8h hi = *(const v8h*)(p + 16);   // K = ko+16 .. ko+23
  v16h f;
#pragma unroll
  for (int i = 0; i < 8; ++i) { f[i] = lo[i]; f[i + 8] = hi[i]; }
  return f;
}

#define WMMA_F16(A, B, C) \
  __builtin_amdgcn_wmma_f32_16x16x32_f16(false, (A), false, (B), (short)0, (C), false, false)

// packed f32x2 -> f16x2 -> one 32-bit LDS store
__device__ __forceinline__ void pack_store(_Float16* dst, float a, float b) {
  union { _Float16 h[2]; unsigned int u; } pk;
  pk.h[0] = (_Float16)a; pk.h[1] = (_Float16)b;
  *(unsigned int*)dst = pk.u;
}

// ---------------------------------------------------------------------------
// Projection GEMM: Out[n,h,l,d] (f16, head-major) = X[4096,1024] @ W[1024,1024] + b
// fp32 inputs converted to f16 (packed) while filling LDS. Tile 128x64, BK=64.
// ---------------------------------------------------------------------------
__global__ __launch_bounds__(256) void proj_gemm_kernel(
    const float* __restrict__ X, const float* __restrict__ W,
    const float* __restrict__ bias, _Float16* __restrict__ Out) {
  __shared__ __align__(16) _Float16 lsA[128][72];
  __shared__ __align__(16) _Float16 lsB[64][72];
  const int t = threadIdx.x, lane = t & 31, wid = t >> 5;
  const int wm = wid >> 1, wn = wid & 1;          // 4x2 wave grid
  const int m0 = blockIdx.y * 128;
  const int n0 = blockIdx.x * 64;

  v8f acc[2][2] = {};
  for (int kb = 0; kb < DM; kb += 64) {
#pragma unroll
    for (int i = 0; i < 16; ++i) {                // A: 128x64 f32 -> f16 pairs
      int idx = i * 256 + t; int r = idx >> 5, c2 = (idx & 31) * 2;
      const float* px = &X[(size_t)(m0 + r) * DM + kb + c2];
      pack_store(&lsA[r][c2], px[0], px[1]);
    }
#pragma unroll
    for (int i = 0; i < 8; ++i) {                 // B: W[k][n] -> lsB[n][k] pairs
      int idx = i * 256 + t; int n = idx & 63, k2 = (idx >> 6) * 2;
      float w0 = W[(size_t)(kb + k2) * DM + n0 + n];
      float w1 = W[(size_t)(kb + k2 + 1) * DM + n0 + n];
      pack_store(&lsB[n][k2], w0, w1);
    }
    if (kb + 64 < DM) {                           // prefetch next K-tile
      __builtin_prefetch(&X[(size_t)(m0 + (t >> 1)) * DM + kb + 64 + (t & 1) * 32], 0, 1);
      __builtin_prefetch(&W[(size_t)(kb + 64 + (t >> 2)) * DM + n0 + (t & 3) * 16], 0, 1);
    }
    __syncthreads();
#pragma unroll
    for (int ks = 0; ks < 64; ks += 32) {
      v16h b0 = lds_frag(&lsB[0][0], 72, wn * 32 + 0,  ks);
      v16h b1 = lds_frag(&lsB[0][0], 72, wn * 32 + 16, ks);
#pragma unroll
      for (int mt = 0; mt < 2; ++mt) {
        v16h a = lds_frag(&lsA[0][0], 72, wm * 32 + mt * 16, ks);
        acc[mt][0] = WMMA_F16(a, b0, acc[mt][0]);
        acc[mt][1] = WMMA_F16(a, b1, acc[mt][1]);
      }
    }
    __syncthreads();
  }
#pragma unroll
  for (int mt = 0; mt < 2; ++mt)
#pragma unroll
    for (int nt = 0; nt < 2; ++nt) {
      int col = n0 + wn * 32 + nt * 16 + (lane & 15);
      int h = col >> 6, d = col & 63;
      float bi = bias[col];
#pragma unroll
      for (int j = 0; j < 8; ++j) {
        int row = m0 + wm * 32 + mt * 16 + ((lane & 16) ? 8 + j : j);
        int nb = row >> 11, l = row & (SEQ - 1);
        Out[(((size_t)nb * NHD + h) * SEQ + l) * DH + d] =
            (_Float16)(acc[mt][nt][j] + bi);
      }
    }
}

// ---------------------------------------------------------------------------
// Column softmax statistics. One block owns a 64-wide s-strip of one (n,h)
// head and sweeps all L tiles: pass 1 -> column max, pass 2 -> column sumexp.
// Fully deterministic block-local LDS reductions (no atomics).
// ---------------------------------------------------------------------------
__global__ __launch_bounds__(256) void colstats_kernel(
    const _Float16* __restrict__ Q, const _Float16* __restrict__ K,
    float* __restrict__ colmax, float* __restrict__ colsum) {
  __shared__ __align__(16) _Float16 lsQ[128][72];
  __shared__ __align__(16) _Float16 lsK[64][72];
  __shared__ float red[8][64];
  __shared__ float cmF[64];
  const int t = threadIdx.x, lane = t & 31, wid = t >> 5;
  const int wm = wid >> 1, wn = wid & 1;
  const int nh = blockIdx.y, s0 = blockIdx.x * 64;
  const _Float16* Qh = Q + (size_t)nh * SEQ * DH;
  const _Float16* Kh = K + (size_t)nh * SEQ * DH;
  const int rrow = wm * 2 + (lane >> 4);
  const int rc   = wn * 32 + (lane & 15);

#pragma unroll
  for (int i = 0; i < 2; ++i) {                   // K strip resident (b128 copies)
    int idx = i * 256 + t; int s = idx >> 3, cc = (idx & 7) * 8;
    *(v8h*)&lsK[s][cc] = *(const v8h*)&Kh[(size_t)(s0 + s) * DH + cc];
  }

  // ---- pass 1: column max ----
  float cm0 = -3.0e38f, cm1 = -3.0e38f;
  for (int lt = 0; lt < SEQ / 128; ++lt) {
    __syncthreads();
#pragma unroll
    for (int i = 0; i < 4; ++i) {                 // Q tile (b128 copies)
      int idx = i * 256 + t; int r = idx >> 3, cc = (idx & 7) * 8;
      *(v8h*)&lsQ[r][cc] = *(const v8h*)&Qh[(size_t)(lt * 128 + r) * DH + cc];
    }
    if (lt + 1 < SEQ / 128)
      __builtin_prefetch(&Qh[(size_t)((lt + 1) * 128 + (t >> 1)) * DH + (t & 1) * 32], 0, 1);
    __syncthreads();
    v8f acc[2][2] = {};
#pragma unroll
    for (int kb = 0; kb < DH; kb += 32) {
      v16h b0 = lds_frag(&lsK[0][0], 72, wn * 32 + 0,  kb);
      v16h b1 = lds_frag(&lsK[0][0], 72, wn * 32 + 16, kb);
#pragma unroll
      for (int mt = 0; mt < 2; ++mt) {
        v16h a = lds_frag(&lsQ[0][0], 72, wm * 32 + mt * 16, kb);
        acc[mt][0] = WMMA_F16(a, b0, acc[mt][0]);
        acc[mt][1] = WMMA_F16(a, b1, acc[mt][1]);
      }
    }
#pragma unroll
    for (int mt = 0; mt < 2; ++mt)
#pragma unroll
      for (int j = 0; j < 8; ++j) {
        cm0 = fmaxf(cm0, acc[mt][0][j] * SCORE_SCALE);
        cm1 = fmaxf(cm1, acc[mt][1][j] * SCORE_SCALE);
      }
  }
  red[rrow][rc] = cm0;
  red[rrow][rc + 16] = cm1;
  __syncthreads();
  if (t < 64) {
    float v = red[0][t];
#pragma unroll
    for (int r = 1; r < 8; ++r) v = fmaxf(v, red[r][t]);
    colmax[(size_t)nh * SEQ + s0 + t] = v;
    cmF[t] = v;
  }
  __syncthreads();
  const float m0v = cmF[wn * 32 + (lane & 15)];
  const float m1v = cmF[wn * 32 + 16 + (lane & 15)];

  // ---- pass 2: column sum of exp ----
  float cs0 = 0.f, cs1 = 0.f;
  for (int lt = 0; lt < SEQ / 128; ++lt) {
    __syncthreads();
#pragma unroll
    for (int i = 0; i < 4; ++i) {
      int idx = i * 256 + t; int r = idx >> 3, cc = (idx & 7) * 8;
      *(v8h*)&lsQ[r][cc] = *(const v8h*)&Qh[(size_t)(lt * 128 + r) * DH + cc];
    }
    if (lt + 1 < SEQ / 128)
      __builtin_prefetch(&Qh[(size_t)((lt + 1) * 128 + (t >> 1)) * DH + (t & 1) * 32], 0, 1);
    __syncthreads();
    v8f acc[2][2] = {};
#pragma unroll
    for (int kb = 0; kb < DH; kb += 32) {
      v16h b0 = lds_frag(&lsK[0][0], 72, wn * 32 + 0,  kb);
      v16h b1 = lds_frag(&lsK[0][0], 72, wn * 32 + 16, kb);
#pragma unroll
      for (int mt = 0; mt < 2; ++mt) {
        v16h a = lds_frag(&lsQ[0][0], 72, wm * 32 + mt * 16, kb);
        acc[mt][0] = WMMA_F16(a, b0, acc[mt][0]);
        acc[mt][1] = WMMA_F16(a, b1, acc[mt][1]);
      }
    }
#pragma unroll
    for (int mt = 0; mt < 2; ++mt)
#pragma unroll
      for (int j = 0; j < 8; ++j) {
        cs0 += __expf(acc[mt][0][j] * SCORE_SCALE - m0v);
        cs1 += __expf(acc[mt][1][j] * SCORE_SCALE - m1v);
      }
  }
  __syncthreads();
  red[rrow][rc] = cs0;
  red[rrow][rc + 16] = cs1;
  __syncthreads();
  if (t < 64) {
    float v = red[0][t];
#pragma unroll
    for (int r = 1; r < 8; ++r) v += red[r][t];
    colsum[(size_t)nh * SEQ + s0 + t] = v;
  }
}

// ---------------------------------------------------------------------------
// attn = softmax(QK^T/8, axis=query) @ V. One block owns a 128-row l-strip of
// one head; loops s tiles of 64: score WMMA -> exp/normalize -> P to LDS ->
// PV WMMA with transposed V tile. O accumulators persist in VGPRs.
// ---------------------------------------------------------------------------
__global__ __launch_bounds__(256) void attn_pv_kernel(
    const _Float16* __restrict__ Q, const _Float16* __restrict__ K,
    const _Float16* __restrict__ V, const float* __restrict__ colmax,
    const float* __restrict__ colsum, _Float16* __restrict__ AOut) {
  __shared__ __align__(16) _Float16 lsQ[128][72];
  __shared__ __align__(16) _Float16 lsK[64][72];
  __shared__ __align__(16) _Float16 lsV[64][72];   // transposed: [d][s]
  __shared__ __align__(16) _Float16 lsP[128][72];  // [l][s]
  const int t = threadIdx.x, lane = t & 31, wid = t >> 5;
  const int wm = wid >> 1, wn = wid & 1;
  const int nh = blockIdx.y, l0 = blockIdx.x * 128;
  const int nb = nh >> 4, hh = nh & 15;
  const _Float16* Qh = Q + (size_t)nh * SEQ * DH;
  const _Float16* Kh = K + (size_t)nh * SEQ * DH;
  const _Float16* Vh = V + (size_t)nh * SEQ * DH;
  const float* cmax = colmax + (size_t)nh * SEQ;
  const float* csum = colsum + (size_t)nh * SEQ;

#pragma unroll
  for (int i = 0; i < 4; ++i) {                   // Q strip resident (b128)
    int idx = i * 256 + t; int r = idx >> 3, cc = (idx & 7) * 8;
    *(v8h*)&lsQ[r][cc] = *(const v8h*)&Qh[(size_t)(l0 + r) * DH + cc];
  }
  v8f oacc[2][2] = {};
  for (int st = 0; st < SEQ / 64; ++st) {
    const int s0 = st * 64;
    __syncthreads();
#pragma unroll
    for (int i = 0; i < 2; ++i) {                 // K tile (b128 copies)
      int idx = i * 256 + t; int s = idx >> 3, cc = (idx & 7) * 8;
      *(v8h*)&lsK[s][cc] = *(const v8h*)&Kh[(size_t)(s0 + s) * DH + cc];
    }
#pragma unroll
    for (int i = 0; i < 2; ++i) {                 // V tile transposed into LDS
      int idx = i * 256 + t; int s = idx >> 3, d0 = (idx & 7) * 8;
      v8h vv = *(const v8h*)&Vh[(size_t)(s0 + s) * DH + d0];
#pragma unroll
      for (int j = 0; j < 8; ++j) lsV[d0 + j][s] = vv[j];
    }
    if (st + 1 < SEQ / 64) {                      // prefetch next s-tile
      __builtin_prefetch(&Kh[(size_t)(s0 + 64 + (t >> 2)) * DH + (t & 3) * 16], 0, 1);
      __builtin_prefetch(&Vh[(size_t)(s0 + 64 + (t >> 2)) * DH + (t & 3) * 16], 0, 1);
    }
    __syncthreads();
    // scores: 128(l) x 64(s)
    v8f sacc[2][2] = {};
#pragma unroll
    for (int kb = 0; kb < DH; kb += 32) {
      v16h b0 = lds_frag(&lsK[0][0], 72, wn * 32 + 0,  kb);
      v16h b1 = lds_frag(&lsK[0][0], 72, wn * 32 + 16, kb);
#pragma unroll
      for (int mt = 0; mt < 2; ++mt) {
        v16h a = lds_frag(&lsQ[0][0], 72, wm * 32 + mt * 16, kb);
        sacc[mt][0] = WMMA_F16(a, b0, sacc[mt][0]);
        sacc[mt][1] = WMMA_F16(a, b1, sacc[mt][1]);
      }
    }
    // P = exp(score*scale - m[s]) / Z[s]  (lane -> column, VGPR -> row)
    const int c0 = wn * 32 + (lane & 15);
    const float m0v = cmax[s0 + c0];
    const float m1v = cmax[s0 + c0 + 16];
    const float rz0 = 1.f / csum[s0 + c0];
    const float rz1 = 1.f / csum[s0 + c0 + 16];
#pragma unroll
    for (int mt = 0; mt < 2; ++mt)
#pragma unroll
      for (int j = 0; j < 8; ++j) {
        int row = wm * 32 + mt * 16 + ((lane & 16) ? 8 + j : j);
        lsP[row][c0]      = (_Float16)(__expf(sacc[mt][0][j] * SCORE_SCALE - m0v) * rz0);
        lsP[row][c0 + 16] = (_Float16)(__expf(sacc[mt][1][j] * SCORE_SCALE - m1v) * rz1);
      }
    __syncthreads();
    // O += P @ Vtile  (k-dim = s = 64)
#pragma unroll
    for (int kb = 0; kb < 64; kb += 32) {
      v16h b0 = lds_frag(&lsV[0][0], 72, wn * 32 + 0,  kb);
      v16h b1 = lds_frag(&lsV[0][0], 72, wn * 32 + 16, kb);
#pragma unroll
      for (int mt = 0; mt < 2; ++mt) {
        v16h a = lds_frag(&lsP[0][0], 72, wm * 32 + mt * 16, kb);
        oacc[mt][0] = WMMA_F16(a, b0, oacc[mt][0]);
        oacc[mt][1] = WMMA_F16(a, b1, oacc[mt][1]);
      }
    }
  }
  // store attn in [n][l][h*64+d] layout (f16) for the output GEMM
#pragma unroll
  for (int mt = 0; mt < 2; ++mt)
#pragma unroll
    for (int nt = 0; nt < 2; ++nt) {
      int col = hh * DH + wn * 32 + nt * 16 + (lane & 15);
#pragma unroll
      for (int j = 0; j < 8; ++j) {
        int l = l0 + wm * 32 + mt * 16 + ((lane & 16) ? 8 + j : j);
        AOut[((size_t)nb * SEQ + l) * DM + col] = (_Float16)oacc[mt][nt][j];
      }
    }
}

// ---------------------------------------------------------------------------
// Output GEMM: out[4096,1024] (f32) = attn[4096,1024] (f16) @ Wo + bo
// Tile 128x64, BK=64, f16 A copied with b128 vector moves.
// ---------------------------------------------------------------------------
__global__ __launch_bounds__(256) void out_gemm_kernel(
    const _Float16* __restrict__ A, const float* __restrict__ W,
    const float* __restrict__ bias, float* __restrict__ Out) {
  __shared__ __align__(16) _Float16 lsA[128][72];
  __shared__ __align__(16) _Float16 lsB[64][72];
  const int t = threadIdx.x, lane = t & 31, wid = t >> 5;
  const int wm = wid >> 1, wn = wid & 1;
  const int m0 = blockIdx.y * 128;
  const int n0 = blockIdx.x * 64;

  v8f acc[2][2] = {};
  for (int kb = 0; kb < DM; kb += 64) {
#pragma unroll
    for (int i = 0; i < 4; ++i) {                 // A: f16 b128 copies
      int idx = i * 256 + t; int r = idx >> 3, cc = (idx & 7) * 8;
      *(v8h*)&lsA[r][cc] = *(const v8h*)&A[(size_t)(m0 + r) * DM + kb + cc];
    }
#pragma unroll
    for (int i = 0; i < 8; ++i) {                 // B: W[k][n] -> lsB[n][k] pairs
      int idx = i * 256 + t; int n = idx & 63, k2 = (idx >> 6) * 2;
      float w0 = W[(size_t)(kb + k2) * DM + n0 + n];
      float w1 = W[(size_t)(kb + k2 + 1) * DM + n0 + n];
      pack_store(&lsB[n][k2], w0, w1);
    }
    if (kb + 64 < DM) {
      __builtin_prefetch(&A[(size_t)(m0 + (t >> 1)) * DM + kb + 64 + (t & 1) * 32], 0, 1);
      __builtin_prefetch(&W[(size_t)(kb + 64 + (t >> 2)) * DM + n0 + (t & 3) * 16], 0, 1);
    }
    __syncthreads();
#pragma unroll
    for (int ks = 0; ks < 64; ks += 32) {
      v16h b0 = lds_frag(&lsB[0][0], 72, wn * 32 + 0,  ks);
      v16h b1 = lds_frag(&lsB[0][0], 72, wn * 32 + 16, ks);
#pragma unroll
      for (int mt = 0; mt < 2; ++mt) {
        v16h a = lds_frag(&lsA[0][0], 72, wm * 32 + mt * 16, ks);
        acc[mt][0] = WMMA_F16(a, b0, acc[mt][0]);
        acc[mt][1] = WMMA_F16(a, b1, acc[mt][1]);
      }
    }
    __syncthreads();
  }
#pragma unroll
  for (int mt = 0; mt < 2; ++mt)
#pragma unroll
    for (int nt = 0; nt < 2; ++nt) {
      int col = n0 + wn * 32 + nt * 16 + (lane & 15);
      float bi = bias[col];
#pragma unroll
      for (int j = 0; j < 8; ++j) {
        int row = m0 + wm * 32 + mt * 16 + ((lane & 16) ? 8 + j : j);
        Out[(size_t)row * DM + col] = acc[mt][nt][j] + bi;
      }
    }
}

// ---------------------------------------------------------------------------
extern "C" void kernel_launch(void* const* d_in, const int* in_sizes, int n_in,
                              void* d_out, int out_size, void* d_ws, size_t ws_size,
                              hipStream_t stream) {
  (void)in_sizes; (void)n_in; (void)out_size; (void)ws_size;
  const float* queries = (const float*)d_in[0];
  const float* keys    = (const float*)d_in[1];
  const float* values  = (const float*)d_in[2];
  const float* Wq = (const float*)d_in[3];
  const float* bq = (const float*)d_in[4];
  const float* Wk = (const float*)d_in[5];
  const float* bk = (const float*)d_in[6];
  const float* Wv = (const float*)d_in[7];
  const float* bv = (const float*)d_in[8];
  const float* Wo = (const float*)d_in[9];
  const float* bo = (const float*)d_in[10];
  float* out = (float*)d_out;

  // Workspace carve-up (~33 MB total)
  const size_t headElems = (size_t)NH * SEQ * DH;      // 4,194,304 halfs each
  _Float16* Qh = (_Float16*)d_ws;
  _Float16* Kh = Qh + headElems;
  _Float16* Vh = Kh + headElems;
  _Float16* At = Vh + headElems;                       // [4096][1024] f16
  float* colmax = (float*)(At + (size_t)MTOT * DM);    // [NH][SEQ]
  float* colsum = colmax + (size_t)NH * SEQ;

  dim3 blk(256);
  dim3 gGemm(DM / 64, MTOT / 128);      // 16 x 32
  dim3 gStat(SEQ / 64, NH);             // 32 x 32
  dim3 gAttn(SEQ / 128, NH);            // 16 x 32

  proj_gemm_kernel<<<gGemm, blk, 0, stream>>>(queries, Wq, bq, Qh);
  proj_gemm_kernel<<<gGemm, blk, 0, stream>>>(keys,    Wk, bk, Kh);
  proj_gemm_kernel<<<gGemm, blk, 0, stream>>>(values,  Wv, bv, Vh);
  colstats_kernel<<<gStat, blk, 0, stream>>>(Qh, Kh, colmax, colsum);
  attn_pv_kernel<<<gAttn, blk, 0, stream>>>(Qh, Kh, Vh, colmax, colsum, At);
  out_gemm_kernel<<<gGemm, blk, 0, stream>>>(At, Wo, bo, out);
}
